// SAGEEMB_66898410602744
// MI455X (gfx1250) — compile-verified, matrix operand order checked by reading, not run on hardware
//
#include <hip/hip_runtime.h>
#include <hip/hip_bf16.h>

// ---------------------------------------------------------------------------
// GraphSAGE 3-layer forward for MI455X (gfx1250, wave32, WMMA).
//
//   * h kept in bf16 end-to-end (GEMM rounds to bf16 anyway): halves the
//     L2 gather traffic of the dominant edge-aggregation phase and halves
//     GEMM A-tile bytes. Accumulation (hn) stays fp32 via L2 atomics.
//   * Self A-tile is DMA'd into LDS with global_load_async_to_lds_b128
//     (ASYNCcnt) while the neighbor tile is scale+converted by VALU.
//   * GEMM: v_wmma_f32_16x16x32_bf16, 16 waves/block, fused bias+ReLU.
// ---------------------------------------------------------------------------

typedef __attribute__((ext_vector_type(16))) __bf16 v16bf;
typedef __attribute__((ext_vector_type(8)))  __bf16 v8bf;
typedef __attribute__((ext_vector_type(8)))  float  v8f;

#define N_NODES 50000
#define N_EDGES 800000
#define D_IN    128
#define D_HID   256

static __device__ __forceinline__ unsigned short f2bf(float f) {
    union { float f; unsigned u; } c; c.f = f;
    unsigned u = c.u;
    unsigned r = u + 0x7FFFu + ((u >> 16) & 1u);   // round-to-nearest-even
    return (unsigned short)(r >> 16);
}

static __device__ __forceinline__ float bfbits_lo(unsigned u) {
    union { unsigned u; float f; } c; c.u = u << 16; return c.f;
}
static __device__ __forceinline__ float bfbits_hi(unsigned u) {
    union { unsigned u; float f; } c; c.u = u & 0xFFFF0000u; return c.f;
}

static __device__ __forceinline__ v16bf ldfrag(const unsigned short* p) {
    // ISA 7.12.2: per-lane 16-bit A/B fragment = K {base..base+7} and
    // {base+16..base+23}; two aligned 16-byte loads, concatenated.
    v8bf lo = *(const v8bf*)(p);
    v8bf hi = *(const v8bf*)(p + 16);
    return __builtin_shufflevector(lo, hi, 0,1,2,3,4,5,6,7,8,9,10,11,12,13,14,15);
}

// ---------------------------------------------------------------------------
__global__ void zero_f(float* __restrict__ p, int n) {
    int i = blockIdx.x * blockDim.x + threadIdx.x;
    if (i < n) p[i] = 0.0f;
}

__global__ void count_deg(const int* __restrict__ dst, float* __restrict__ deg, int ne) {
    int e = blockIdx.x * blockDim.x + threadIdx.x;
    if (e < ne) atomicAdd(&deg[dst[e]], 1.0f);
}

__global__ void recip_deg(float* __restrict__ deg, int n) {
    int i = blockIdx.x * blockDim.x + threadIdx.x;
    if (i < n) deg[i] = 1.0f / fmaxf(deg[i], 1.0f);
}

__global__ void f32_to_bf16(const float* __restrict__ in,
                            unsigned short* __restrict__ out, int n) {
    int i = blockIdx.x * blockDim.x + threadIdx.x;
    if (i < n) out[i] = f2bf(in[i]);
}

// W[K x N] fp32 -> Wt[N x K] bf16 bits (row = output column, contiguous in K)
__global__ void wconvT(const float* __restrict__ W, unsigned short* __restrict__ Wt,
                       int Kd, int Nd) {
    int i = blockIdx.x * blockDim.x + threadIdx.x;
    if (i < Kd * Nd) {
        int k = i / Nd;
        int n = i - k * Nd;
        Wt[n * Kd + k] = f2bf(W[i]);
    }
}

// Edge-parallel gather (bf16, 16B = 8 features per lane) + fp32 scatter-add.
// d/8 consecutive lanes per edge -> one wave covers a full 256-wide row with
// a single coalesced 512B read; atomics land in the L2-resident hn buffer.
__global__ void scatter_add_bf(const unsigned short* __restrict__ h,
                               const int* __restrict__ src,
                               const int* __restrict__ dst,
                               float* __restrict__ hn, int ne, int d) {
    int idx = blockIdx.x * blockDim.x + threadIdx.x;
    int chunks = d >> 3;
    int e = idx / chunks;
    if (e >= ne) return;
    int c = (idx - e * chunks) << 3;
    const uint4 v = *(const uint4*)(h + (size_t)src[e] * d + c);
    float* o = hn + (size_t)dst[e] * d + c;
    atomicAdd(o + 0, bfbits_lo(v.x)); atomicAdd(o + 1, bfbits_hi(v.x));
    atomicAdd(o + 2, bfbits_lo(v.y)); atomicAdd(o + 3, bfbits_hi(v.y));
    atomicAdd(o + 4, bfbits_lo(v.z)); atomicAdd(o + 5, bfbits_hi(v.z));
    atomicAdd(o + 6, bfbits_lo(v.w)); atomicAdd(o + 7, bfbits_hi(v.w));
}

// ---------------------------------------------------------------------------
// Fused SAGE layer GEMM: out = relu(As @ Ws + (An*inv_deg) @ Wn + b)
// Block: 512 threads = 16 waves; one 16-row M-tile; wave w owns cols
// [16w,16w+16). Self tile async-DMA'd bf16->LDS; neighbor tile converted.
// ---------------------------------------------------------------------------
template<int K>
__global__ __launch_bounds__(512)
void sage_gemm(const unsigned short* __restrict__ Asb,   // [M x K] bf16 self
               const float* __restrict__ An,             // [M x K] fp32 neigh sums
               const float* __restrict__ invdeg,
               const unsigned short* __restrict__ WsT,   // [256 x K] bf16
               const unsigned short* __restrict__ WnT,   // [256 x K] bf16
               const float* __restrict__ bias,
               unsigned short* __restrict__ outb,        // [M x 256] bf16 (next layer)
               float* __restrict__ outf) {               // [M x 256] fp32 or nullptr
    constexpr int LDA = K + 8;          // +16B pad: spreads rows across LDS banks
    constexpr int NB  = 16 * LDA;
    __shared__ __align__(16) unsigned short sA[2 * NB];

    const int tid = threadIdx.x;
    const int m0  = blockIdx.x * 16;

    // Self tile: async DMA bf16 global -> LDS, 16B (8 elems) per lane.
    // CH is a multiple of 32, so the guard is whole-wave uniform.
    {
        constexpr int CH = (16 * K) / 8;           // 256 (K=128) or 512 (K=256)
        if (tid < CH) {
            const int r = tid / (K / 8);
            const int c = (tid - r * (K / 8)) * 8;
            // Low 32 bits of a flat pointer into LDS == LDS byte offset
            // (shared aperture lives entirely in the high 32 bits).
            unsigned lds_addr = (unsigned)(size_t)(&sA[r * LDA + c]);
            unsigned goff     = (unsigned)(((unsigned)(m0 + r) * K + c) * 2u);
            asm volatile("global_load_async_to_lds_b128 %0, %1, %2"
                         :: "v"(lds_addr), "v"(goff), "s"(Asb)
                         : "memory");
        }
    }

    // Neighbor tile: fp32 load, fused mean (inv_deg), bf16 convert, LDS store.
    for (int idx = tid; idx < 16 * K; idx += 512) {
        const int r = idx / K;
        const int c = idx - r * K;
        sA[NB + r * LDA + c] = f2bf(An[(size_t)(m0 + r) * K + c] * invdeg[m0 + r]);
    }

    asm volatile("s_wait_asynccnt 0x0" ::: "memory");
    __syncthreads();

    const int lane = tid & 31;
    const int wave = tid >> 5;            // 0..15 -> N-tile
    const int nl   = lane & 15;           // A row (M) / B row (N) per ISA layout
    const int kb   = (lane >= 16) ? 8 : 0;
    const int ncol = wave * 16 + nl;

    const unsigned short* arow_s = sA + nl * LDA + kb;
    const unsigned short* arow_n = sA + NB + nl * LDA + kb;
    const unsigned short* brow_s = WsT + (size_t)ncol * K + kb;
    const unsigned short* brow_n = WnT + (size_t)ncol * K + kb;

    v8f acc;
    const float bv = bias[ncol];
    #pragma unroll
    for (int r = 0; r < 8; ++r) acc[r] = bv;

    #pragma unroll
    for (int k0 = 0; k0 < K; k0 += 32) {
        v16bf a_s = ldfrag(arow_s + k0);
        v16bf b_s = ldfrag(brow_s + k0);
        acc = __builtin_amdgcn_wmma_f32_16x16x32_bf16(
                  false, a_s, false, b_s, (short)0, acc, false, false);
        v16bf a_n = ldfrag(arow_n + k0);
        v16bf b_n = ldfrag(brow_n + k0);
        acc = __builtin_amdgcn_wmma_f32_16x16x32_bf16(
                  false, a_n, false, b_n, (short)0, acc, false, false);
    }

    // C/D layout: VGPR r -> row (r | r+8 by lane half), col = lane&15.
    const int mbase = m0 + ((lane >= 16) ? 8 : 0);
    #pragma unroll
    for (int r = 0; r < 8; ++r) {
        const float v = fmaxf(acc[r], 0.0f);
        outb[(size_t)(mbase + r) * D_HID + ncol] = f2bf(v);
    }
    if (outf) {                                   // uniform branch (final layer)
        #pragma unroll
        for (int r = 0; r < 8; ++r) {
            outf[(size_t)(mbase + r) * D_HID + ncol] = fmaxf(acc[r], 0.0f);
        }
    }
}

// ---------------------------------------------------------------------------
extern "C" void kernel_launch(void* const* d_in, const int* in_sizes, int n_in,
                              void* d_out, int out_size, void* d_ws, size_t ws_size,
                              hipStream_t stream) {
    const float* x   = (const float*)d_in[0];
    const int*   src = (const int*)  d_in[1];
    const int*   dst = (const int*)  d_in[2];
    const float* Ws0 = (const float*)d_in[3];
    const float* Wn0 = (const float*)d_in[4];
    const float* b0  = (const float*)d_in[5];
    const float* Ws1 = (const float*)d_in[6];
    const float* Wn1 = (const float*)d_in[7];
    const float* b1  = (const float*)d_in[8];
    const float* Ws2 = (const float*)d_in[9];
    const float* Wn2 = (const float*)d_in[10];
    const float* b2  = (const float*)d_in[11];
    float* out = (float*)d_out;

    // Workspace carve (256B aligned): ~117 MB total.
    char* w = (char*)d_ws;
    auto carve = [&](size_t bytes) -> void* {
        void* p = (void*)w;
        w += (bytes + 255) & ~(size_t)255;
        return p;
    };
    float*          invdeg = (float*)         carve((size_t)N_NODES * 4);
    float*          hn     = (float*)         carve((size_t)N_NODES * D_HID * 4);
    unsigned short* xb     = (unsigned short*)carve((size_t)N_NODES * D_IN  * 2);
    unsigned short* hb1    = (unsigned short*)carve((size_t)N_NODES * D_HID * 2);
    unsigned short* hb2    = (unsigned short*)carve((size_t)N_NODES * D_HID * 2);
    unsigned short* Wt0s   = (unsigned short*)carve((size_t)D_IN  * D_HID * 2);
    unsigned short* Wt0n   = (unsigned short*)carve((size_t)D_IN  * D_HID * 2);
    unsigned short* Wt1s   = (unsigned short*)carve((size_t)D_HID * D_HID * 2);
    unsigned short* Wt1n   = (unsigned short*)carve((size_t)D_HID * D_HID * 2);
    unsigned short* Wt2s   = (unsigned short*)carve((size_t)D_HID * D_HID * 2);
    unsigned short* Wt2n   = (unsigned short*)carve((size_t)D_HID * D_HID * 2);

    const int T = 256;
    auto blks = [](long long n, int t) { return (unsigned)((n + t - 1) / t); };

    // --- inverse in-degree (shared by all layers) ---
    zero_f   <<<blks(N_NODES, T), T, 0, stream>>>(invdeg, N_NODES);
    count_deg<<<blks(N_EDGES, T), T, 0, stream>>>(dst, invdeg, N_EDGES);
    recip_deg<<<blks(N_NODES, T), T, 0, stream>>>(invdeg, N_NODES);

    // --- input features + weights -> bf16 (weights transposed) ---
    f32_to_bf16<<<blks((long long)N_NODES * D_IN, T), T, 0, stream>>>(x, xb, N_NODES * D_IN);
    wconvT<<<blks((long long)D_IN  * D_HID, T), T, 0, stream>>>(Ws0, Wt0s, D_IN,  D_HID);
    wconvT<<<blks((long long)D_IN  * D_HID, T), T, 0, stream>>>(Wn0, Wt0n, D_IN,  D_HID);
    wconvT<<<blks((long long)D_HID * D_HID, T), T, 0, stream>>>(Ws1, Wt1s, D_HID, D_HID);
    wconvT<<<blks((long long)D_HID * D_HID, T), T, 0, stream>>>(Wn1, Wt1n, D_HID, D_HID);
    wconvT<<<blks((long long)D_HID * D_HID, T), T, 0, stream>>>(Ws2, Wt2s, D_HID, D_HID);
    wconvT<<<blks((long long)D_HID * D_HID, T), T, 0, stream>>>(Wn2, Wt2n, D_HID, D_HID);

    const unsigned MB = N_NODES / 16;   // 3125, exact

    // --- layer 0 (K = 128) ---
    zero_f<<<blks((long long)N_NODES * D_IN, T), T, 0, stream>>>(hn, N_NODES * D_IN);
    scatter_add_bf<<<blks((long long)N_EDGES * (D_IN / 8), T), T, 0, stream>>>(
        xb, src, dst, hn, N_EDGES, D_IN);
    sage_gemm<D_IN><<<MB, 512, 0, stream>>>(xb, hn, invdeg, Wt0s, Wt0n, b0,
                                            hb1, (float*)nullptr);

    // --- layer 1 (K = 256) ---
    zero_f<<<blks((long long)N_NODES * D_HID, T), T, 0, stream>>>(hn, N_NODES * D_HID);
    scatter_add_bf<<<blks((long long)N_EDGES * (D_HID / 8), T), T, 0, stream>>>(
        hb1, src, dst, hn, N_EDGES, D_HID);
    sage_gemm<D_HID><<<MB, 512, 0, stream>>>(hb1, hn, invdeg, Wt1s, Wt1n, b1,
                                             hb2, (float*)nullptr);

    // --- layer 2 (K = 256), fp32 result to d_out ---
    zero_f<<<blks((long long)N_NODES * D_HID, T), T, 0, stream>>>(hn, N_NODES * D_HID);
    scatter_add_bf<<<blks((long long)N_EDGES * (D_HID / 8), T), T, 0, stream>>>(
        hb2, src, dst, hn, N_EDGES, D_HID);
    sage_gemm<D_HID><<<MB, 512, 0, stream>>>(hb2, hn, invdeg, Wt2s, Wt2n, b2,
                                             hb1, out);
}
